// C2FAB_Heads_55353538511541
// MI455X (gfx1250) — matrix operand order repaired and checked by reading, not compiled, not run on hardware
//
#include <hip/hip_runtime.h>
#include <math.h>
#include <stdint.h>

#define H     4096
#define HID   256
#define S_U   4096
#define S_Q   1024
#define BATCH 4
#define DDIM  8
#define MTILE 64

typedef __attribute__((ext_vector_type(16))) __bf16 bf16x16;
typedef __attribute__((ext_vector_type(8)))  __bf16 bf16x8;
typedef __attribute__((ext_vector_type(8)))  float  f32x8;
typedef __attribute__((ext_vector_type(4)))  unsigned int u32x4;
typedef __attribute__((ext_vector_type(8)))  int   i32x8;
typedef __attribute__((ext_vector_type(4)))  int   i32x4;

union BF16V { bf16x16 v; bf16x8 h[2]; };

#if __has_builtin(__builtin_amdgcn_tensor_load_to_lds) && __has_builtin(__builtin_amdgcn_s_wait_tensorcnt)
#define HAVE_TDM 1
#else
#define HAVE_TDM 0
#endif

__device__ __forceinline__ float gelu_exact(float x) {
    return 0.5f * x * (1.0f + erff(x * 0.70710678118654752f));
}

#if HAVE_TDM
// TDM: DMA a 1-D f32 array (len elements, len < 65536) from global into LDS.
// Descriptor per CDNA5 ISA; 6-arg builtin on this toolchain.
__device__ __forceinline__ void tdm_load_1d_f32(const float* gsrc, void* lds_dst, int len)
{
    unsigned long long ga = (unsigned long long)(uintptr_t)gsrc;
    unsigned lds = (unsigned)(uintptr_t)lds_dst;      // low 32 bits = LDS byte offset

    u32x4 g0;
    g0[0] = 1u;                                        // count=1, is_restore=0, gather off
    g0[1] = lds;                                       // lds_addr
    g0[2] = (unsigned)(ga & 0xFFFFFFFFu);              // global_addr[31:0]
    g0[3] = (unsigned)((ga >> 32) & 0x1FFFFFFu)        // global_addr[56:32]
          | (2u << 30);                                // type=2 (image)

    i32x8 g1;
    g1[0] = (int)(2u << 16);                           // workgroup_mask=0, data_size=4B
    g1[1] = (int)(((unsigned)len & 0xFFFFu) << 16);    // tensor_dim0[15:0] in bits[63:48]
    g1[2] = (int)((((unsigned)len >> 16) & 0xFFFFu)    // tensor_dim0[31:16]
          | (1u << 16));                               // tensor_dim1[15:0] = 1
    g1[3] = (int)(((unsigned)len & 0xFFFFu) << 16);    // tensor_dim1 hi=0, tile_dim0=len
    g1[4] = 1;                                         // tile_dim1=1, tile_dim2=0
    g1[5] = len;                                       // tensor_dim0_stride[31:0]
    g1[6] = 0;
    g1[7] = 0;

    i32x4 g2 = {0, 0, 0, 0};
    i32x4 g3 = {0, 0, 0, 0};
    i32x8 g4 = {0, 0, 0, 0, 0, 0, 0, 0};
    __builtin_amdgcn_tensor_load_to_lds(g0, g1, g2, g3, g4, 0);
}
#endif

// B fragment: column n of W chunk, K = kbase..kbase+15 (bf16).
__device__ __forceinline__ bf16x16 load_bfrag(const __bf16* Wt, const float* Wf,
                                              int n, int kbase)
{
    if (Wt) {
        return *(const bf16x16*)(Wt + (size_t)n * H + kbase);
    } else {
        union { bf16x16 v; __bf16 e[16]; } u;
        #pragma unroll
        for (int i = 0; i < 16; ++i)
            u.e[i] = (__bf16)Wf[(size_t)(kbase + i) * HID + n];
        return u.v;
    }
}

// One block: 64 rows of LN -> (x @ Wa + ba) -> GELU -> (h @ Wb + bb) [-> ReLU]
template <int NOUT, bool RELU>
__global__ __launch_bounds__(256)
void ln_mlp_wmma_kernel(const float* __restrict__ x,
                        const float* __restrict__ ln_g, const float* __restrict__ ln_b,
                        const float* __restrict__ Wa_f32,
                        const __bf16* __restrict__ Wt,      // [HID][H] bf16, may be null
                        const float* __restrict__ ba,
                        const float* __restrict__ Wb, const float* __restrict__ bb,
                        float* __restrict__ out)
{
    __shared__ float  s_mu[MTILE], s_rs[MTILE];
    __shared__ float  s_red[2][256];
    __shared__ float  sG[H];                 // ln gamma (TDM-loaded)
    __shared__ float  sBln[H];               // ln beta  (TDM-loaded)
    __shared__ __bf16 sA[MTILE * 32];        // LN-applied A chunk, row-major [m][k]
    __shared__ float  sH[MTILE * 256];       // GELU(h) for second GEMM

    const int t    = threadIdx.x;
    const int row0 = blockIdx.x * MTILE;

    // ---- LN params -> LDS (TDM DMA if available) ----
#if HAVE_TDM
    if (t == 0) {
        tdm_load_1d_f32(ln_g, &sG[0], H);
        tdm_load_1d_f32(ln_b, &sBln[0], H);
        __builtin_amdgcn_s_wait_tensorcnt(0);
    }
#else
    for (int k = t; k < H; k += 256) { sG[k] = ln_g[k]; sBln[k] = ln_b[k]; }
#endif

    // ---- LayerNorm statistics: 4 threads per row, float4 vectorized ----
    {
        int r = t >> 2, l4 = t & 3;
        const float4* xr4 = (const float4*)(x + (size_t)(row0 + r) * H);
        float s = 0.f, q = 0.f;
        for (int k = l4; k < H / 4; k += 4) {
            float4 v = xr4[k];
            s += v.x + v.y + v.z + v.w;
            q += v.x * v.x + v.y * v.y + v.z * v.z + v.w * v.w;
        }
        s_red[0][t] = s; s_red[1][t] = q;
    }
    __syncthreads();
    if (t < MTILE) {
        float s = 0.f, q = 0.f;
        #pragma unroll
        for (int i = 0; i < 4; ++i) { s += s_red[0][t * 4 + i]; q += s_red[1][t * 4 + i]; }
        float mu  = s * (1.0f / H);
        float var = q * (1.0f / H) - mu * mu;
        s_mu[t] = mu;
        s_rs[t] = rsqrtf(var + 1e-5f);
    }
    __syncthreads();

    const int wid   = t >> 5;
    const int lane  = t & 31;
    const int m     = lane & 15;
    const int half8 = (lane < 16) ? 0 : 8;    // C row base
    const int bhalf = (lane < 16) ? 0 : 16;   // B K-half base
    const int n0    = wid * 32;               // this wave's 2 column tiles

    // Stage-A mapping: thread owns 8 contiguous elements of one row
    const int sr  = t >> 2;           // row 0..63
    const int skb = (t & 3) * 8;      // col base 0/8/16/24
    const float mu_r = s_mu[sr];      // loop-invariant: hoisted into registers
    const float rs_r = s_rs[sr];

    f32x8 c[4][2];
    #pragma unroll
    for (int i = 0; i < 4; ++i)
        #pragma unroll
        for (int j = 0; j < 2; ++j)
            c[i][j] = (f32x8){0.f, 0.f, 0.f, 0.f, 0.f, 0.f, 0.f, 0.f};

    // One K-chunk (32 wide): stage A, load frags, prefetch next B into bn*, 8 WMMAs on bu*.
    auto chunk = [&](int k0, bf16x16& bu0, bf16x16& bu1,
                     bf16x16& bn0, bf16x16& bn1, bool more) {
        __syncthreads();   // previous chunk's fragment reads done before restage
        {
            const float* xp = x + (size_t)(row0 + sr) * H + k0 + skb;
            float4 xa = *(const float4*)xp;
            float4 xb = *(const float4*)(xp + 4);
            float4 ga = *(const float4*)&sG[k0 + skb];
            float4 gb = *(const float4*)&sG[k0 + skb + 4];
            float4 ta = *(const float4*)&sBln[k0 + skb];
            float4 tb = *(const float4*)&sBln[k0 + skb + 4];
            union { bf16x8 v; __bf16 e[8]; } pk;
            pk.e[0] = (__bf16)((xa.x - mu_r) * rs_r * ga.x + ta.x);
            pk.e[1] = (__bf16)((xa.y - mu_r) * rs_r * ga.y + ta.y);
            pk.e[2] = (__bf16)((xa.z - mu_r) * rs_r * ga.z + ta.z);
            pk.e[3] = (__bf16)((xa.w - mu_r) * rs_r * ga.w + ta.w);
            pk.e[4] = (__bf16)((xb.x - mu_r) * rs_r * gb.x + tb.x);
            pk.e[5] = (__bf16)((xb.y - mu_r) * rs_r * gb.y + tb.y);
            pk.e[6] = (__bf16)((xb.z - mu_r) * rs_r * gb.z + tb.z);
            pk.e[7] = (__bf16)((xb.w - mu_r) * rs_r * gb.w + tb.w);
            *(bf16x8*)&sA[sr * 32 + skb] = pk.v;
            if (more) __builtin_prefetch(xp + 32, 0, 1);
        }
        __syncthreads();

        // A fragments: lanes 0-15 row m K=0..15 ; lanes 16-31 K=16..31
        BF16V a[4];
        #pragma unroll
        for (int i = 0; i < 4; ++i) {
            a[i].h[0] = *(const bf16x8*)&sA[(i * 16 + m) * 32 + bhalf];
            a[i].h[1] = *(const bf16x8*)&sA[(i * 16 + m) * 32 + bhalf + 8];
        }

        if (more) {
            bn0 = load_bfrag(Wt, Wa_f32, n0 + m,      k0 + 32 + bhalf);
            bn1 = load_bfrag(Wt, Wa_f32, n0 + 16 + m, k0 + 32 + bhalf);
        }

        #pragma unroll
        for (int i = 0; i < 4; ++i) {
            c[i][0] = __builtin_amdgcn_wmma_f32_16x16x32_bf16(false, a[i].v, false, bu0,
                                                              (short)0, c[i][0], false, false);
            c[i][1] = __builtin_amdgcn_wmma_f32_16x16x32_bf16(false, a[i].v, false, bu1,
                                                              (short)0, c[i][1], false, false);
        }
    };

    // Ping-pong register double-buffer of B fragments (no copies, H/32 is even)
    bf16x16 p0 = load_bfrag(Wt, Wa_f32, n0 + m,      bhalf);
    bf16x16 p1 = load_bfrag(Wt, Wa_f32, n0 + 16 + m, bhalf);
    bf16x16 q0 = p0, q1 = p1;
    for (int k0 = 0; k0 < H; k0 += 64) {
        chunk(k0,      p0, p1, q0, q1, true);
        chunk(k0 + 32, q0, q1, p0, p1, (k0 + 64) < H);
    }
    __syncthreads();

    // Epilogue 1: bias + exact GELU into sH.
    // C layout: VGPR v, lanes 0-15 -> (row v, col lane); lanes 16-31 -> (row v+8, col lane-16)
    #pragma unroll
    for (int i = 0; i < 4; ++i) {
        #pragma unroll
        for (int j = 0; j < 2; ++j) {
            int ncol = n0 + j * 16 + m;
            #pragma unroll
            for (int v = 0; v < 8; ++v) {
                int rr = i * 16 + v + half8;
                sH[rr * 256 + ncol] = gelu_exact(c[i][j][v] + ba[ncol]);
            }
        }
    }
    __syncthreads();

    // Epilogue 2: tiny second GEMM [64 x HID] @ [HID x NOUT]
    for (int o = t; o < MTILE * NOUT; o += 256) {
        int r = o / NOUT, cc = o % NOUT;
        float acc = bb[cc];
        const float* hr = &sH[r * 256];
        #pragma unroll 8
        for (int k = 0; k < HID; ++k) acc += hr[k] * Wb[k * NOUT + cc];
        if (RELU) acc = fmaxf(acc, 0.f);
        out[(size_t)(row0 + r) * NOUT + cc] = acc;
    }
}

// Transpose-convert W [H][HID] f32 -> Wt [HID][H] bf16 (coalesced both sides)
__global__ __launch_bounds__(256)
void wt_convert_kernel(const float* __restrict__ W, __bf16* __restrict__ Wt)
{
    __shared__ float tile[32][33];
    int k0 = blockIdx.x * 32;           // gridDim.x = H/32
    int n0 = blockIdx.y * 32;           // gridDim.y = HID/32
    int r = threadIdx.x >> 5, cidx = threadIdx.x & 31;
    #pragma unroll
    for (int i = 0; i < 32; i += 8)
        tile[r + i][cidx] = W[(size_t)(k0 + r + i) * HID + n0 + cidx];
    __syncthreads();
    #pragma unroll
    for (int i = 0; i < 32; i += 8)
        Wt[(size_t)(n0 + r + i) * H + k0 + cidx] = (__bf16)tile[cidx][r + i];
}

// One wave per (batch, lambda, d): fwd + bwd IIR over S=4096 via chunked scan
// + cross-lane shuffle scan of (lam^128, carry) pairs.
__global__ __launch_bounds__(32)
void iir_phi_kernel(const float* __restrict__ Cu,
                    const float* __restrict__ lam_fast,
                    const float* __restrict__ lam_slow,
                    float* __restrict__ Phi)
{
    const int blk    = blockIdx.x;        // 64 blocks
    const int d      = blk & 7;
    const int lamsel = (blk >> 3) & 1;
    const int b      = blk >> 4;
    const int lane   = threadIdx.x;

    float lam = lamsel ? lam_slow[d] : lam_fast[d];
    lam = fminf(fmaxf(lam, 1e-4f), 1.0f - 1e-4f);

    const float* xs = Cu  + (size_t)b * S_U * DDIM + d;                // stride DDIM
    float*       ph = Phi + (size_t)b * S_U * 16 + lamsel * DDIM + d;  // stride 16

    float P = lam;
    #pragma unroll
    for (int i = 0; i < 7; ++i) P *= P;   // lam^128

    const int CH = S_U / 32;              // 128 elements per lane

    for (int dir = 0; dir < 2; ++dir) {
        float L = 0.f;
        for (int j = 0; j < CH; ++j) {
            int tt  = lane * CH + j;
            int pos = dir ? (S_U - 1 - tt) : tt;
            L = lam * L + xs[(size_t)pos * DDIM];
        }
        float a = P, v = L;
        #pragma unroll
        for (int off = 1; off < 32; off <<= 1) {
            float v2 = __shfl_up(v, off, 32);
            float a2 = __shfl_up(a, off, 32);
            if (lane >= off) { v = v2 * a + v; a = a2 * a; }
        }
        float cin = __shfl_up(v, 1, 32);
        if (lane == 0) cin = 0.f;
        float y = cin;
        for (int j = 0; j < CH; ++j) {
            int tt  = lane * CH + j;
            int pos = dir ? (S_U - 1 - tt) : tt;
            y = lam * y + xs[(size_t)pos * DDIM];
            if (dir == 0) ph[(size_t)pos * 16] = y;
            else          ph[(size_t)pos * 16] += y;
        }
    }
}

extern "C" void kernel_launch(void* const* d_in, const int* in_sizes, int n_in,
                              void* d_out, int out_size, void* d_ws, size_t ws_size,
                              hipStream_t stream) {
    const float* x_u      = (const float*)d_in[0];
    const float* x_q      = (const float*)d_in[1];
    const float* ln1_g    = (const float*)d_in[2];
    const float* ln1_b    = (const float*)d_in[3];
    const float* W1       = (const float*)d_in[4];
    const float* b1       = (const float*)d_in[5];
    const float* W2       = (const float*)d_in[6];
    const float* b2       = (const float*)d_in[7];
    const float* ln2_g    = (const float*)d_in[8];
    const float* ln2_b    = (const float*)d_in[9];
    const float* W3       = (const float*)d_in[10];
    const float* b3       = (const float*)d_in[11];
    const float* W4       = (const float*)d_in[12];
    const float* b4       = (const float*)d_in[13];
    const float* lam_fast = (const float*)d_in[14];
    const float* lam_slow = (const float*)d_in[15];

    float* out = (float*)d_out;
    float* Phi = out;                                   // [4,4096,16]
    float* Rq  = out + (size_t)BATCH * S_U * 16;        // [4,1024,16]
    float* Cu  = Rq  + (size_t)BATCH * S_Q * 16;        // [4,4096,8]

    // Pre-convert W1/W3 to column-major bf16 in workspace (if it fits)
    const size_t wt_elems = (size_t)H * HID;
    __bf16* Wt1 = nullptr; __bf16* Wt3 = nullptr;
    if (ws_size >= 2 * wt_elems * sizeof(__bf16)) {
        Wt1 = (__bf16*)d_ws;
        Wt3 = Wt1 + wt_elems;
        dim3 g(H / 32, HID / 32);
        wt_convert_kernel<<<g, 256, 0, stream>>>(W1, Wt1);
        wt_convert_kernel<<<g, 256, 0, stream>>>(W3, Wt3);
    }

    ln_mlp_wmma_kernel<DDIM, true><<<(BATCH * S_U) / MTILE, 256, 0, stream>>>(
        x_u, ln1_g, ln1_b, W1, Wt1, b1, W2, b2, Cu);

    ln_mlp_wmma_kernel<2 * DDIM, false><<<(BATCH * S_Q) / MTILE, 256, 0, stream>>>(
        x_q, ln2_g, ln2_b, W3, Wt3, b3, W4, b4, Rq);

    iir_phi_kernel<<<BATCH * 2 * DDIM, 32, 0, stream>>>(Cu, lam_fast, lam_slow, Phi);
}
